// FLIF_B_74972949119028
// MI455X (gfx1250) — matrix operand order repaired and checked by reading, not compiled, not run on hardware
//
#include <hip/hip_runtime.h>

// ---------------------------------------------------------------------------
// FLIF recurrent SNN forward on MI455X (gfx1250, wave32, WMMA).
//
// Phase A: split Wi/Wr f32 -> (f16 hi, f16 lo residual); error-free f16 GEMM
//          decomposition: f32 accum of hi*hi + hi*lo + lo*hi recovers ~f32
//          accuracy (dropped lo*lo term ~2^-22) on the fast f16 WMMA pipe.
// Phase B: I_in = x @ Wi.T as a device-wide LDS-tiled WMMA GEMM
//          (1024 workgroups of 8 waves, 128x128 output tiles). Output is
//          written in WMMA C-fragment-major layout so the scan can reload it
//          with coalesced b128s instead of per-element scalar loads.
// Phase C: 256-step scan. Batch is the only dependence-free axis; WMMA M=16
//          => 8 independent workgroups (16 batches each, all 512 outputs),
//          spikes double-buffered in LDS (exact in f16), Wr hi/lo streamed
//          from the 192MB L2 every step, one s_barrier per step.
// ---------------------------------------------------------------------------

typedef __attribute__((ext_vector_type(16))) _Float16 v16h;
typedef __attribute__((ext_vector_type(8)))  _Float16 v8h;
typedef __attribute__((ext_vector_type(4)))  _Float16 v4h;
typedef __attribute__((ext_vector_type(8)))  float    v8f;

namespace {
constexpr int T_STEPS = 256;
constexpr int BATCH   = 128;
constexpr int IN_DIM  = 512;
constexpr int OUT_DIM = 512;
constexpr int M1      = T_STEPS * BATCH;          // 32768 GEMM rows

constexpr float ALPHA   = 0.81873075307798182f;   // exp(-dt/i_tau) = exp(-0.2)
constexpr float BETA    = 0.90483741803595957f;   // exp(-dt/v_tau) = exp(-0.1)
constexpr float THR_TAU = 0.99f;
}  // namespace

__device__ __forceinline__ v16h mk16(v8h lo, v8h hi) {
  union { v16h v; v8h h[2]; } u;
  u.h[0] = lo; u.h[1] = hi;
  return u.v;
}

__device__ __forceinline__ v8f wmma_f16(v16h a, v16h b, v8f c) {
  // (neg_a, A, neg_b, B, c_mod, C, reuse_a, reuse_b)
  return __builtin_amdgcn_wmma_f32_16x16x32_f16(false, a, false, b, (short)0, c,
                                                false, false);
}

// f32 -> (f16 hi, f16 residual lo) split
__global__ __launch_bounds__(256) void split_f16_kernel(
    const float* __restrict__ src, _Float16* __restrict__ hi,
    _Float16* __restrict__ lo, int n) {
  int i = blockIdx.x * 256 + threadIdx.x;
  if (i < n) {
    float v = src[i];
    _Float16 h = (_Float16)v;
    hi[i] = h;
    lo[i] = (_Float16)(v - (float)h);
  }
}

// I_in[m, n] = sum_k x[m, k] * Wi[n, k]   (m = t*B + b; one 128-row block
// per blockIdx.x is exactly one timestep t with all 128 batches)
// Output layout: fragment-major iin[t][bt(8)][nt(32)][lane(32)][v(8)]
__global__ __launch_bounds__(256) void flif_input_gemm(
    const float*    __restrict__ x,    // [M1, IN] f32 (split to hi/lo in LDS)
    const _Float16* __restrict__ wih,  // [OUT, IN] f16 hi
    const _Float16* __restrict__ wil,  // [OUT, IN] f16 residual
    float*          __restrict__ iin)  // frag-major, M1*OUT floats
{
  __shared__ __align__(16) _Float16 Axh[128 * 32];
  __shared__ __align__(16) _Float16 Axl[128 * 32];
  __shared__ __align__(16) _Float16 Bwh[128 * 32];
  __shared__ __align__(16) _Float16 Bwl[128 * 32];

  const int tid  = threadIdx.x;
  const int wave = tid >> 5;               // 8 waves, each a 16x128 strip
  const int lane = tid & 31;
  const int lr   = lane & 15;
  const int ka   = (lane < 16) ? 0 : 16;   // A frag K sub-offset (contiguous k = 16h + s)
  const int bks  = (lane < 16) ? 0 : 16;   // B frag K sub-offset (contiguous k = 16h + s)

  const int m0 = blockIdx.x * 128;
  const int n0 = blockIdx.y * 128;

  v8f acc[8] = {};

  for (int kb = 0; kb < IN_DIM; kb += 32) {
    __syncthreads();
    // Stage x tile with on-the-fly hi/lo split: 128 rows x 32 k
    for (int j = tid; j < 128 * 8; j += 256) {
      int r = j >> 3, seg = j & 7;
      float4 f =
          ((const float4*)x)[(size_t)(m0 + r) * (IN_DIM / 4) + (kb >> 2) + seg];
      int base = r * 32 + seg * 4;
      v4h h, l;
      h[0] = (_Float16)f.x; h[1] = (_Float16)f.y;
      h[2] = (_Float16)f.z; h[3] = (_Float16)f.w;
      l[0] = (_Float16)(f.x - (float)h[0]);
      l[1] = (_Float16)(f.y - (float)h[1]);
      l[2] = (_Float16)(f.z - (float)h[2]);
      l[3] = (_Float16)(f.w - (float)h[3]);
      *(v4h*)(Axh + base) = h;
      *(v4h*)(Axl + base) = l;
    }
    // Stage Wi hi/lo tiles: 128 rows (n) x 32 k, already f16 in global
    for (int j = tid; j < 512; j += 256) {
      int r = j >> 2, seg = j & 3;
      ((uint4*)Bwh)[j] =
          ((const uint4*)wih)[(size_t)(n0 + r) * (IN_DIM / 8) + (kb >> 3) + seg];
      ((uint4*)Bwl)[j] =
          ((const uint4*)wil)[(size_t)(n0 + r) * (IN_DIM / 8) + (kb >> 3) + seg];
    }
    __syncthreads();

    const int arow = wave * 16 + lr;
    v16h ah = mk16(*(const v8h*)(Axh + arow * 32 + ka),
                   *(const v8h*)(Axh + arow * 32 + ka + 8));
    v16h al = mk16(*(const v8h*)(Axl + arow * 32 + ka),
                   *(const v8h*)(Axl + arow * 32 + ka + 8));
#pragma unroll
    for (int ns = 0; ns < 8; ++ns) {
      int bcol = ns * 16 + lr;
      v16h bh = mk16(*(const v8h*)(Bwh + bcol * 32 + bks),
                     *(const v8h*)(Bwh + bcol * 32 + bks + 8));
      v16h bl = mk16(*(const v8h*)(Bwl + bcol * 32 + bks),
                     *(const v8h*)(Bwl + bcol * 32 + bks + 8));
      acc[ns] = wmma_f16(ah, bh, acc[ns]);  // hi*hi
      acc[ns] = wmma_f16(ah, bl, acc[ns]);  // hi*lo
      acc[ns] = wmma_f16(al, bh, acc[ns]);  // lo*hi  (lo*lo ~2^-22, dropped)
    }
  }

  // Fragment-major store: t = blockIdx.x, bt = wave, nt = blockIdx.y*8 + ns.
  // Each lane writes its 8 accumulator floats contiguously (2x b128).
  float* dst = iin +
               (((size_t)blockIdx.x * 8 + wave) * 32 + (size_t)blockIdx.y * 8) *
                   256 +
               (size_t)lane * 8;
#pragma unroll
  for (int ns = 0; ns < 8; ++ns)
    *(v8f*)(dst + (size_t)ns * 256) = acc[ns];
}

// 256-step scan; one workgroup owns 16 batches x all 512 outputs.
__global__ __launch_bounds__(512) void flif_scan(
    const float*    __restrict__ iin,        // frag-major (see gemm)
    const _Float16* __restrict__ wrh,        // [OUT, OUT] f16 hi
    const _Float16* __restrict__ wrl,        // [OUT, OUT] f16 residual
    const float*    __restrict__ thr,        // [OUT]
    const float*    __restrict__ thr_delta,  // [OUT]
    float*          __restrict__ out)        // [T, B, OUT] f32 spikes
{
  __shared__ __align__(16) _Float16 spk[2][16 * OUT_DIM];  // double-buffered

  const int tid    = threadIdx.x;
  const int wave   = tid >> 5;  // 16 waves; wave owns cols [32w, 32w+32)
  const int lane   = tid & 31;
  const int lr     = lane & 15;
  const int ka     = (lane < 16) ? 0 : 16;
  const int bks    = (lane < 16) ? 0 : 16;
  const int rowoff = (lane >> 4) * 8;

  const int b0 = blockIdx.x * 16;  // this workgroup's batch slice

  for (int j = tid; j < 16 * OUT_DIM / 2; j += 512)
    ((unsigned*)spk[0])[j] = 0u;  // spk_0 = 0

  float colthr[2], coltd[2];
  float V[2][8], Ic[2][8], TE[2][8];
#pragma unroll
  for (int tt = 0; tt < 2; ++tt) {
    int col = wave * 32 + tt * 16 + lr;
    colthr[tt] = thr[col];
    coltd[tt]  = thr_delta[col];
#pragma unroll
    for (int v = 0; v < 8; ++v) {
      V[tt][v] = 0.f; Ic[tt][v] = 0.f; TE[tt][v] = colthr[tt];
    }
  }
  __syncthreads();

  for (int t = 0; t < T_STEPS; ++t) {
    const _Float16* sp = spk[t & 1];
    _Float16* sn = spk[(t + 1) & 1];

    // Issue this step's I_in fragment loads early: 2x v8f per lane (4x b128),
    // latency hidden under the 16-chunk recurrent WMMA pipeline below.
    const float* ip = iin +
                      (((size_t)t * 8 + blockIdx.x) * 32 + (size_t)wave * 2) *
                          256 +
                      (size_t)lane * 8;
    v8f iinf[2];
    iinf[0] = *(const v8f*)ip;
    iinf[1] = *(const v8f*)(ip + 256);

    // rec[b, n] = sum_k spk[b, k] * Wr[n, k]; spikes exact in f16, Wr hi+lo
    v8f acc[2] = {};
    for (int kb = 0; kb < OUT_DIM; kb += 32) {
      v16h a = mk16(*(const v8h*)(sp + lr * OUT_DIM + kb + ka),
                    *(const v8h*)(sp + lr * OUT_DIM + kb + ka + 8));
#pragma unroll
      for (int tt = 0; tt < 2; ++tt) {
        const size_t nrow =
            (size_t)(wave * 32 + tt * 16 + lr) * OUT_DIM + kb + bks;
        if (kb + 32 < OUT_DIM) __builtin_prefetch(wrh + nrow + 32, 0, 0);
        v16h bh = mk16(*(const v8h*)(wrh + nrow), *(const v8h*)(wrh + nrow + 8));
        v16h bl = mk16(*(const v8h*)(wrl + nrow), *(const v8h*)(wrl + nrow + 8));
        acc[tt] = wmma_f16(a, bh, acc[tt]);
        acc[tt] = wmma_f16(a, bl, acc[tt]);
      }
    }

    const size_t trow = (size_t)t * BATCH;
#pragma unroll
    for (int tt = 0; tt < 2; ++tt) {
      const int col = wave * 32 + tt * 16 + lr;
#pragma unroll
      for (int v = 0; v < 8; ++v) {
        const int brow = rowoff + v;
        const size_t gidx = (trow + b0 + brow) * OUT_DIM + col;
        float Iv = ALPHA * Ic[tt][v] + iinf[tt][v] + acc[tt][v];
        float Vv = BETA * V[tt][v] + Iv;
        float te = TE[tt][v];
        float s  = (Vv - te) > 0.f ? 1.f : 0.f;
        Vv -= s * te;
        te = THR_TAU * (te - colthr[tt]) + colthr[tt] + coltd[tt] * s;
        Ic[tt][v] = Iv; V[tt][v] = Vv; TE[tt][v] = te;
        out[gidx] = s;
        sn[brow * OUT_DIM + col] = (_Float16)s;  // exact {0,1}
      }
    }
    __syncthreads();  // one barrier per step: WAR-safe via double buffer
  }
}

extern "C" void kernel_launch(void* const* d_in, const int* in_sizes, int n_in,
                              void* d_out, int out_size, void* d_ws,
                              size_t ws_size, hipStream_t stream) {
  (void)in_sizes; (void)n_in; (void)out_size; (void)ws_size;
  const float* x   = (const float*)d_in[0];
  const float* Wi  = (const float*)d_in[1];
  const float* Wr  = (const float*)d_in[2];
  const float* thr = (const float*)d_in[3];
  const float* td  = (const float*)d_in[4];
  float* out = (float*)d_out;

  // Workspace layout (66 MiB total):
  //   [0, 64MiB)        I_in_all f32, fragment-major [t][bt][nt][lane][v]
  //   [64MiB, +512KiB)  Wi hi f16, then Wi lo f16
  //   [65MiB, +512KiB)  Wr hi f16, then Wr lo f16
  char* ws = (char*)d_ws;
  float*    iin = (float*)ws;
  _Float16* wih = (_Float16*)(ws + (64ull << 20));
  _Float16* wil = (_Float16*)(ws + (64ull << 20) + (512u << 10));
  _Float16* wrh = (_Float16*)(ws + (65ull << 20));
  _Float16* wrl = (_Float16*)(ws + (65ull << 20) + (512u << 10));

  const int wn = OUT_DIM * IN_DIM;  // 262144
  split_f16_kernel<<<(wn + 255) / 256, 256, 0, stream>>>(Wi, wih, wil, wn);
  split_f16_kernel<<<(wn + 255) / 256, 256, 0, stream>>>(Wr, wrh, wrl, wn);

  dim3 g1(M1 / 128, OUT_DIM / 128);  // 256 x 4 workgroups
  flif_input_gemm<<<g1, 256, 0, stream>>>(x, wih, wil, iin);

  flif_scan<<<BATCH / 16, 512, 0, stream>>>(iin, wrh, wrl, thr, td, out);
}
